// ST_GCNN_TorsoHeart_15075335209079
// MI455X (gfx1250) — matrix operand order repaired and checked by reading, not compiled, not run on hardware
//
#include <hip/hip_runtime.h>

typedef _Float16 v16h __attribute__((ext_vector_type(16)));
typedef _Float16 v8h  __attribute__((ext_vector_type(8)));
typedef float    v8f  __attribute__((ext_vector_type(8)));
typedef unsigned int v4u __attribute__((ext_vector_type(4)));
typedef int      v4i __attribute__((ext_vector_type(4)));
typedef int      v8i __attribute__((ext_vector_type(8)));

#define NB    32           // batch
#define NT    256          // torso nodes
#define NHRT  512          // heart nodes
#define FDIM  64           // features
#define SDIM  25           // lattice
#define GDIM  (NB * SDIM)  // 800 GEMM rows
#define NE    2048         // edges
#define MTILES 50          // 800/16
#define CHUNKS 5           // 50 tiles / 10 waves

// workspace layout (bytes)
#define WS_OFFS   0                         // 513 ints
#define WS_SRCS   4096                      // 2048 ints: src torso node per sorted pos
#define WS_INV    12288                     // 2048 ints: sorted pos per edge
#define WS_FRAGS  24576                     // 2048 * 8192 B f16 B-fragments (sorted order)
#define WS_XT     (24576 + (size_t)NE * 8192)   // 800*256*64 f16

static __device__ __forceinline__ v16h combine16(v8h lo, v8h hi) {
  v16h r;
#pragma unroll
  for (int i = 0; i < 8; ++i) { r[i] = lo[i]; r[8 + i] = hi[i]; }
  return r;
}

// ---------------- zero output ----------------
__global__ __launch_bounds__(256) void k_zero(float4* __restrict__ out) {
  size_t i = (size_t)blockIdx.x * 256 + threadIdx.x;
  out[i] = make_float4(0.f, 0.f, 0.f, 0.f);
}

// ------- deterministic group-by-dst (stable order, no atomics), gather srcs, inverse map -------
__global__ __launch_bounds__(512) void k_group(const int* __restrict__ dst,
                                               const int* __restrict__ srcI,
                                               int* __restrict__ offs,
                                               int* __restrict__ srcs,
                                               int* __restrict__ inv) {
  __shared__ int scnt[NHRT];
  __shared__ int soff[NHRT + 1];
  int d = threadIdx.x;
  int cnt = 0;
  for (int e = 0; e < NE; ++e) cnt += (dst[e] == d) ? 1 : 0;
  scnt[d] = cnt;
  __syncthreads();
  if (d == 0) {
    int run = 0;
    for (int i = 0; i < NHRT; ++i) { soff[i] = run; run += scnt[i]; }
    soff[NHRT] = run;
  }
  __syncthreads();
  int pos = soff[d];
  for (int e = 0; e < NE; ++e)
    if (dst[e] == d) {
      srcs[pos] = srcI[e] - NHRT;
      inv[e] = pos;
      pos++;
    }
  offs[d] = soff[d];
  if (d == 0) offs[NHRT] = soff[NHRT];
}

// ---------------- z (B,NT,F,S) f32 -> xt (G,NT,F) f16 ----------------
__global__ __launch_bounds__(256) void k_xt(const float* __restrict__ z,
                                            _Float16* __restrict__ xt) {
  unsigned idx = blockIdx.x * 256u + threadIdx.x;   // exact: 800*256*64 = 51200*256
  unsigned f  = idx & 63u;
  unsigned nt = (idx >> 6) & 255u;
  unsigned g  = idx >> 14;
  unsigned b  = g & 31u;
  unsigned s  = g >> 5;
  float v = z[(((size_t)b * NT + nt) * FDIM + f) * SDIM + s];
  xt[idx] = (_Float16)v;
}

// ---------------- W_eff per edge -> WMMA B-fragments (f16), stored at sorted pos ----------------
// B is K(=f_in) x N(=f_out). frag fi = kt*4+nt: K in [kt*32,kt*32+32), N in [nt*16,nt*16+16).
// B 32x16 fragment order used here (contiguous k, same as A): lanes 0-15 K=kt*32+0..15, lanes 16-31 K=kt*32+16..31, n = nt*16+(lane&15).
__device__ __forceinline__ float qbasis(float t, int i) {
  if (i == 0) return 0.5f * (1.f - t) * (1.f - t);
  if (i == 1) return -t * t + t + 0.5f;
  return 0.5f * t * t;
}

__global__ __launch_bounds__(256) void k_weff(const float* __restrict__ ea,
                                              const float* __restrict__ w,
                                              const int* __restrict__ inv,
                                              _Float16* __restrict__ frags) {
  __shared__ float coef[27];
  int e = blockIdx.x;
  if (threadIdx.x < 27) {
    int a = threadIdx.x % 3, b = (threadIdx.x / 3) % 3, c = threadIdx.x / 9;
    coef[threadIdx.x] = qbasis(ea[e * 3 + 0], a) *
                        qbasis(ea[e * 3 + 1], b) *
                        qbasis(ea[e * 3 + 2], c);
  }
  __syncthreads();
  int lane = threadIdx.x & 31, fi = threadIdx.x >> 5;
  int kt = fi >> 2, ntile = fi & 3;
  int n = ntile * 16 + (lane & 15);
  int kbase = kt * 32 + (lane >> 4) * 16;
  float acc[16];
#pragma unroll
  for (int i = 0; i < 16; ++i) acc[i] = 0.f;
  for (int j = 0; j < 27; ++j) {
    float cj = coef[j];
    const float* wj = w + (size_t)j * 4096 + (size_t)kbase * 64 + n;
#pragma unroll
    for (int i = 0; i < 16; ++i) acc[i] = fmaf(cj, wj[(size_t)i * 64], acc[i]);
  }
  v8h lo, hi;
#pragma unroll
  for (int i = 0; i < 8; ++i) { lo[i] = (_Float16)acc[i]; hi[i] = (_Float16)acc[8 + i]; }
  int p = inv[e]; // sorted position
  _Float16* op = frags + ((size_t)(p * 8 + fi) * 32 + lane) * 16;
  *(v8h*)op = lo;
  *(v8h*)(op + 8) = hi;
}

// ---------------- main: (dst, M-chunk) blocks; TDM double-buffered B; WMMA ----------------
__global__ __launch_bounds__(320) void k_msg(const _Float16* __restrict__ xt,
                                             const _Float16* __restrict__ frags,
                                             const int* __restrict__ srcs,
                                             const int* __restrict__ offs,
                                             float* __restrict__ out) {
  __shared__ float4 smem[1024]; // 2 x 8KB double buffer of B-fragments
  int d = blockIdx.x;
  int chunk = blockIdx.y;
  int e0 = offs[d], e1 = offs[d + 1];
  if (e0 == e1) return; // zero-init covers this dst

  int wave = threadIdx.x >> 5, lane = threadIdx.x & 31;
  int tile = chunk * 10 + wave;  // 0..49, every wave active
  int m = lane & 15, lg = lane >> 4;
  int g0 = tile * 16;
  bool issuer = (threadIdx.x < 32); // wave 0 drives the TDM
  unsigned ldsbase = (unsigned)(size_t)(void*)smem;

  // TDM descriptor: 1-D tile, 1024 elements of 8 B = one edge's 8 KB fragment block
  auto issue_tdm = [&](unsigned ldsoff, int ei) {
    unsigned long long ga = (unsigned long long)(size_t)(frags + (size_t)ei * 4096);
    v4u g0v;
    g0v[0] = 1u;                                   // count=1, user descriptor
    g0v[1] = ldsoff;                               // lds_addr
    g0v[2] = (unsigned)(ga & 0xFFFFFFFFu);         // global_addr[31:0]
    g0v[3] = (unsigned)((ga >> 32) & 0x1FFFFFFu) | (2u << 30); // addr[56:32] | type=2
    v8i g1v;
    g1v[0] = (int)(3u << 16);                      // data_size=3 (8 B), mask=0, flags=0
    g1v[1] = (int)(1024u << 16);                   // tensor_dim0[15:0] at bits 63:48
    g1v[2] = 0;                                    // tensor_dim0[31:16]=0, tensor_dim1 lo=0
    g1v[3] = (int)(1024u << 16);                   // tile_dim0=1024 at bits 127:112
    g1v[4] = 0;                                    // tile_dim1=0, tile_dim2=0 (1-D)
    g1v[5] = 1024;                                 // tensor_dim0_stride[31:0]
    g1v[6] = 0;
    g1v[7] = 0;
    v4i zz4; zz4[0] = 0; zz4[1] = 0; zz4[2] = 0; zz4[3] = 0;
    v8i zz8;
#pragma unroll
    for (int i = 0; i < 8; ++i) zz8[i] = 0;
    __builtin_amdgcn_tensor_load_to_lds(g0v, g1v, zz4, zz4, zz8, 0);
  };

  // prologue: stage edge e0 into buffer 0
  if (issuer) {
    issue_tdm(ldsbase, e0);
    __builtin_amdgcn_s_wait_tensorcnt((short)0);
  }
  int ns = srcs[e0];
  __syncthreads();

  v8f acc0 = {}, acc1 = {}, acc2 = {}, acc3 = {};

  for (int ei = e0; ei < e1; ++ei) {
    int cur = (ei - e0) & 1;
    bool havenext = (ei + 1 < e1);
    if (havenext && issuer)
      issue_tdm(ldsbase + (cur ? 0u : 8192u), ei + 1); // fill other buffer async
    int ns_next = havenext ? srcs[ei + 1] : 0;
    if (havenext)
      __builtin_prefetch(xt + ((size_t)(g0 + m) * NT + ns_next) * FDIM, 0, 0);

    // A fragments (16x32 f16): row g0+m, lane half lg holds k = 16*lg .. 16*lg+15 per k-tile
    const _Float16* ab = xt + ((size_t)(g0 + m) * NT + ns) * FDIM;
    v16h a0 = combine16(*(const v8h*)(ab + lg * 16),      *(const v8h*)(ab + 8 + lg * 16));
    v16h a1 = combine16(*(const v8h*)(ab + 32 + lg * 16), *(const v8h*)(ab + 40 + lg * 16));

    const char* lb = (const char*)smem + cur * 8192 + lane * 32;
#define LDB(fi) combine16(*(const v8h*)(lb + (fi) * 1024), *(const v8h*)(lb + (fi) * 1024 + 16))
    v16h b0 = LDB(0), b1 = LDB(1), b2 = LDB(2), b3 = LDB(3);
    v16h b4 = LDB(4), b5 = LDB(5), b6 = LDB(6), b7 = LDB(7);
#undef LDB
    acc0 = __builtin_amdgcn_wmma_f32_16x16x32_f16(false, a0, false, b0, (short)0, acc0, false, false);
    acc1 = __builtin_amdgcn_wmma_f32_16x16x32_f16(false, a0, false, b1, (short)0, acc1, false, false);
    acc2 = __builtin_amdgcn_wmma_f32_16x16x32_f16(false, a0, false, b2, (short)0, acc2, false, false);
    acc3 = __builtin_amdgcn_wmma_f32_16x16x32_f16(false, a0, false, b3, (short)0, acc3, false, false);
    acc0 = __builtin_amdgcn_wmma_f32_16x16x32_f16(false, a1, false, b4, (short)0, acc0, false, false);
    acc1 = __builtin_amdgcn_wmma_f32_16x16x32_f16(false, a1, false, b5, (short)0, acc1, false, false);
    acc2 = __builtin_amdgcn_wmma_f32_16x16x32_f16(false, a1, false, b6, (short)0, acc2, false, false);
    acc3 = __builtin_amdgcn_wmma_f32_16x16x32_f16(false, a1, false, b7, (short)0, acc3, false, false);

    if (havenext && issuer)
      __builtin_amdgcn_s_wait_tensorcnt((short)0);  // next buffer landed
    __syncthreads(); // publish next buffer; all reads of current buffer done
    ns = ns_next;
  }

  // C/D layout: VGPR r -> row M = r + 8*(lane/16), col N = lane%16
#pragma unroll
  for (int r = 0; r < 8; ++r) {
    int g = g0 + r + lg * 8;
    int b = g & 31, s = g >> 5;
    size_t base = ((size_t)b * NHRT + d) * FDIM * SDIM + s;
    out[base + (size_t)(0 * 16 + m) * SDIM] = acc0[r];
    out[base + (size_t)(1 * 16 + m) * SDIM] = acc1[r];
    out[base + (size_t)(2 * 16 + m) * SDIM] = acc2[r];
    out[base + (size_t)(3 * 16 + m) * SDIM] = acc3[r];
  }
}

extern "C" void kernel_launch(void* const* d_in, const int* in_sizes, int n_in,
                              void* d_out, int out_size, void* d_ws, size_t ws_size,
                              hipStream_t stream) {
  (void)in_sizes; (void)n_in; (void)ws_size;
  const float* z    = (const float*)d_in[0];
  const float* ea   = (const float*)d_in[1];
  const float* w    = (const float*)d_in[2];
  const int*   srcI = (const int*)d_in[3];
  const int*   dstI = (const int*)d_in[4];
  float* out = (float*)d_out;
  char*  ws  = (char*)d_ws;

  int*      offs  = (int*)(ws + WS_OFFS);
  int*      srcs  = (int*)(ws + WS_SRCS);
  int*      inv   = (int*)(ws + WS_INV);
  _Float16* frags = (_Float16*)(ws + WS_FRAGS);
  _Float16* xt    = (_Float16*)(ws + WS_XT);

  k_zero<<<out_size / 4 / 256, 256, 0, stream>>>((float4*)out);
  k_group<<<1, 512, 0, stream>>>(dstI, srcI, offs, srcs, inv);
  k_xt<<<(GDIM * NT * FDIM) / 256, 256, 0, stream>>>(z, xt);
  k_weff<<<NE, 256, 0, stream>>>(ea, w, inv, frags);
  k_msg<<<dim3(NHRT, CHUNKS), 320, 0, stream>>>(xt, frags, srcs, offs, out);
}
